// LPSparseMAPFaithfulMatching_14508399526047
// MI455X (gfx1250) — matrix operand order repaired and checked by reading, not compiled, not run on hardware
//
#include <hip/hip_runtime.h>
#include <math.h>

// ---------------------------------------------------------------------------
// Dims (fixed by the reference) + padded pitches (multiples of 32 for K-tiles,
// multiples of 8 elements => 16B alignment for vector loads)
// ---------------------------------------------------------------------------
#define BB   64
#define TT   256
#define EE   300
#define HH   200
#define ENC  400          // 2H
#define G4   800          // 4H
#define REP  1600         // 4*enc

#define EP   320          // padded E
#define HP   224          // padded H
#define ENCP 416          // padded 2H

// Flip to 0 to fall back to VGPR-staged (load->ds_store) copies.
#define USE_ASYNC_LDS 1

typedef __attribute__((ext_vector_type(16))) __bf16 v16bf;
typedef __attribute__((ext_vector_type(8)))  float  v8f;

static __device__ __forceinline__ unsigned short f32_to_bf16(float f) {
  union { float f; unsigned u; } x; x.f = f;
  unsigned r = x.u + 0x7FFFu + ((x.u >> 16) & 1u);   // round-to-nearest-even
  return (unsigned short)(r >> 16);
}
static __device__ __forceinline__ float sigm(float x) {
  return 1.0f / (1.0f + expf(-x));
}

#if USE_ASYNC_LDS
// Generic LDS pointer = {SHARED_BASE[63:32], lds_offset[31:0]} (ISA 10.2),
// so truncation yields the wave-relative LDS byte address the async op wants.
static __device__ __forceinline__ unsigned lds_off32(const void* p) {
  return (unsigned)(unsigned long)p;
}
static __device__ __forceinline__ void async_copy_b128(const void* gsrc, void* ldst) {
  asm volatile("global_load_async_to_lds_b128 %0, %1, off"
               :: "v"(lds_off32(ldst)), "v"((unsigned long)gsrc)
               : "memory");
}
static __device__ __forceinline__ void async_copy_b64(const void* gsrc, void* ldst) {
  asm volatile("global_load_async_to_lds_b64 %0, %1, off"
               :: "v"(lds_off32(ldst)), "v"((unsigned long)gsrc)
               : "memory");
}
static __device__ __forceinline__ void wait_asynccnt0() {
#if defined(__has_builtin) && __has_builtin(__builtin_amdgcn_s_wait_asynccnt)
  __builtin_amdgcn_s_wait_asynccnt(0);
#else
  asm volatile("s_wait_asynccnt 0x0" ::: "memory");
#endif
}
#endif

// ---------------------------------------------------------------------------
// Strided/batched WMMA GEMM, bf16 in / f32 out (+optional bf16 mirror).
//   C[b,m,n] = sum_k A(b,m,k)*B(b,k,n) (+C)(+add)(+bias)(relu)
// AMODE: 0 => A k-contiguous (sa_k==1), 1 => A m-contiguous (sa_m==1)
// BMODE: 0 => B k-contiguous (sb_k==1), 1 => B n-contiguous (sb_n==1)
// Caller guarantees: M mult of 64; padded operand buffers cover full 32-wide
// K/N tiles with ZEROS => no load-side bounds checks. Stores are guarded.
// 256 threads = 8 waves -> 64x32 tile (4x2 of 16x16 WMMA).
// ---------------------------------------------------------------------------
template<int AMODE, int BMODE>
__global__ __launch_bounds__(256) void wmma_gemm_t(
    const unsigned short* __restrict__ A, const unsigned short* __restrict__ B,
    const float* __restrict__ bias, const float* __restrict__ add,
    float* __restrict__ C, unsigned short* __restrict__ Cb16,
    int M, int N, int Kp,
    long sa_m, long sa_k, long a_bs,
    long sb_k, long sb_n, long b_bs,
    long c_row, long c_bs, long add_row,
    int relu, int accum)
{
  const int bb = blockIdx.z;
  const int m0 = blockIdx.y * 64;
  const int n0 = blockIdx.x * 32;
  const unsigned short* Ab = A + (long)bb * a_bs;
  const unsigned short* Bb = B + (long)bb * b_bs;
  float* Cp = C + (long)bb * c_bs;

  __shared__ __align__(16) unsigned short lA[64][40];  // [m][k], 80B pitch
  __shared__ __align__(16) unsigned short lB[32][40];  // [n][k]

  const int tid  = threadIdx.x;
  const int lane = tid & 31;
  const int wave = tid >> 5;
  const int wm   = wave & 3;
  const int wn   = wave >> 2;

  v8f acc;
  #pragma unroll
  for (int i = 0; i < 8; ++i) acc[i] = 0.0f;

  for (int k0 = 0; k0 < Kp; k0 += 32) {
    // ---- stage A tile 64x32 ----
    if (AMODE == 0) {
      int r  = tid >> 2;
      int c8 = (tid & 3) * 8;
      const unsigned short* src = Ab + (long)(m0 + r) * sa_m + (k0 + c8);
#if USE_ASYNC_LDS
      async_copy_b128(src, &lA[r][c8]);
#else
      *(uint4*)&lA[r][c8] = *(const uint4*)src;
#endif
    } else {
      int c  = tid >> 3;          // k-local
      int r8 = (tid & 7) * 8;     // m-local base (stride-1 in memory)
      union { uint4 q; unsigned short s[8]; } u;
      u.q = *(const uint4*)(Ab + (long)(k0 + c) * sa_k + (m0 + r8));
      #pragma unroll
      for (int i = 0; i < 8; ++i) lA[r8 + i][c] = u.s[i];
    }
    // ---- stage B tile 32x32 into [n][k] ----
    if (BMODE == 0) {
      int n  = tid >> 3;
      int c4 = (tid & 7) * 4;
      const unsigned short* src = Bb + (long)(n0 + n) * sb_n + (k0 + c4);
#if USE_ASYNC_LDS
      async_copy_b64(src, &lB[n][c4]);
#else
      *(uint2*)&lB[n][c4] = *(const uint2*)src;
#endif
    } else {
      int c  = tid >> 3;          // k-local
      int n4 = (tid & 7) * 4;     // n-local base (stride-1 in memory)
      union { uint2 q; unsigned short s[4]; } u;
      u.q = *(const uint2*)(Bb + (long)(k0 + c) * sb_k + (n0 + n4));
      #pragma unroll
      for (int i = 0; i < 4; ++i) lB[n4 + i][c] = u.s[i];
    }
#if USE_ASYNC_LDS
    if (AMODE == 0 || BMODE == 0) wait_asynccnt0();
#endif
    __syncthreads();

    // ---- fragments: two b128 LDS reads each ----
    union { v16bf v; uint4 q[2]; } fa, fb;
    {
      int ar = lane & 15;
      int kb = (lane < 16) ? 0 : 8;      // A: K {kb..kb+7, kb+16..kb+23}
      fa.q[0] = *(const uint4*)&lA[wm * 16 + ar][kb];
      fa.q[1] = *(const uint4*)&lA[wm * 16 + ar][kb + 16];
      int bc  = lane & 15;
      int kbb = (lane < 16) ? 0 : 8;     // B: K {kbb..kbb+7, kbb+16..kbb+23}
      fb.q[0] = *(const uint4*)&lB[wn * 16 + bc][kbb];
      fb.q[1] = *(const uint4*)&lB[wn * 16 + bc][kbb + 16];
    }
    acc = __builtin_amdgcn_wmma_f32_16x16x32_bf16(
        false, fa.v, false, fb.v, (short)0, acc, false, false);
    __syncthreads();
  }

  // ---- epilogue: C layout VGPR r -> M=r|r+8 by lane half, N=lane%16 ----
  const int nl = lane & 15;
  const int gn = n0 + wn * 16 + nl;
  #pragma unroll
  for (int r = 0; r < 8; ++r) {
    int ml = (lane < 16) ? r : (r + 8);
    int gm = m0 + wm * 16 + ml;
    if (gm < M && gn < N) {
      float val = acc[r];
      long  ci  = (long)gm * c_row + gn;
      if (accum) val += Cp[ci];
      if (add)   val += add[(long)gm * add_row + gn];
      if (bias)  val += bias[gn];
      if (relu)  val = fmaxf(val, 0.0f);
      Cp[ci] = val;
      if (Cb16) Cb16[(long)bb * c_bs + ci] = f32_to_bf16(val);
    }
  }
}

// ---------------------------------------------------------------------------
// Helper kernels
// ---------------------------------------------------------------------------
__global__ void cast_f32_bf16_k(unsigned short* __restrict__ d,
                                const float* __restrict__ s, long n) {
  long i = (long)blockIdx.x * 256 + threadIdx.x;
  if (i < n) d[i] = f32_to_bf16(s[i]);
}

// dst[Npad][Kpad] (bf16) = src[n*src_ld + src_off + k] for n<Nr,k<Kc else 0
__global__ void cast_pad2d_k(unsigned short* __restrict__ dst,
                             const float* __restrict__ src,
                             int src_ld, int src_off,
                             int Nr, int Kc, int Npad, int Kpad) {
  long i = (long)blockIdx.x * 256 + threadIdx.x;
  if (i >= (long)Npad * Kpad) return;
  int n = (int)(i / Kpad), k = (int)(i % Kpad);
  float v = (n < Nr && k < Kc) ? src[(long)n * src_ld + src_off + k] : 0.0f;
  dst[i] = f32_to_bf16(v);
}

__global__ void fill_u32_k(unsigned int* __restrict__ p, long n) {
  long i = (long)blockIdx.x * 256 + threadIdx.x;
  if (i < n) p[i] = 0u;
}

// gather: e_tm[(t*B+b)*EP + e] = bf16(embed[tok*EE+e]) (time-major, padded)
__global__ void embed_gather_k(unsigned short* __restrict__ dst,
                               const float* __restrict__ embed,
                               const int* __restrict__ tok, long n) {
  long i = (long)blockIdx.x * 256 + threadIdx.x;
  if (i >= n) return;
  int e  = (int)(i % EP);
  long bt = i / EP;
  int b  = (int)(bt % BB);
  int t  = (int)(bt / BB);
  float v = 0.0f;
  if (e < EE) {
    int tk = tok[b * TT + t];
    v = embed[(long)tk * EE + e];
  }
  dst[i] = f32_to_bf16(v);
}

// LSTM gates: g [64,800] -> c update; h -> f32 seq + bf16 state (pitch HP)
__global__ void lstm_gate_k(const float* __restrict__ g,
                            float* __restrict__ c,
                            unsigned short* __restrict__ hbf,
                            float* __restrict__ hseq,
                            long hseq_off, long seq_bstride) {
  int i = blockIdx.x * 256 + threadIdx.x;
  if (i >= BB * HH) return;
  int b = i / HH, j = i % HH;
  const float* gr = g + (long)b * G4;
  float gi = gr[j];
  float gf = gr[HH + j];
  float gg = gr[2 * HH + j];
  float go = gr[3 * HH + j];
  float cn = sigm(gf) * c[i] + sigm(gi) * tanhf(gg);
  float hn = sigm(go) * tanhf(cn);
  c[i] = cn;
  hbf[(long)b * HP + j] = f32_to_bf16(hn);
  hseq[hseq_off + (long)b * seq_bstride + j] = hn;
}

// ---------------------------------------------------------------------------
// Capped-simplex projection of a 256-vector (one workgroup, 256 threads)
// ---------------------------------------------------------------------------
static __device__ float proj_capped_simplex256(float v, int tid) {
  __shared__ float zs[256];
  __shared__ float cz[256];
  __shared__ float red[256];

  red[tid] = fmaxf(v, 0.0f); __syncthreads();
  for (int s = 128; s > 0; s >>= 1) {
    if (tid < s) red[tid] += red[tid + s];
    __syncthreads();
  }
  float ssum = red[0]; __syncthreads();

  zs[tid] = v; __syncthreads();
  for (int k = 2; k <= 256; k <<= 1) {          // bitonic sort, descending
    for (int j = k >> 1; j > 0; j >>= 1) {
      int ixj = tid ^ j;
      if (ixj > tid) {
        float a = zs[tid], b = zs[ixj];
        bool desc = ((tid & k) == 0);
        if (desc ? (a < b) : (a > b)) { zs[tid] = b; zs[ixj] = a; }
      }
      __syncthreads();
    }
  }
  cz[tid] = zs[tid]; __syncthreads();           // inclusive scan
  for (int off = 1; off < 256; off <<= 1) {
    float t = (tid >= off) ? cz[tid - off] : 0.0f;
    __syncthreads();
    cz[tid] += t;
    __syncthreads();
  }
  red[tid] = (1.0f + (float)(tid + 1) * zs[tid] > cz[tid]) ? 1.0f : 0.0f;
  __syncthreads();
  for (int s = 128; s > 0; s >>= 1) {
    if (tid < s) red[tid] += red[tid + s];
    __syncthreads();
  }
  int kmax = (int)red[0];
  if (kmax < 1) kmax = 1;
  float tau = (cz[kmax - 1] - 1.0f) / (float)kmax;
  if (ssum <= 1.0f) tau = 0.0f;
  __syncthreads();
  return fmaxf(v - tau, 0.0f);
}

__global__ __launch_bounds__(256) void dykstra_rows_k(float* __restrict__ p,
                                                      float* __restrict__ q1,
                                                      const float* __restrict__ q2) {
  int i = blockIdx.x, b = blockIdx.y, tid = threadIdx.x;
  long base = ((long)b * TT + i) * TT + tid;
  float v = p[base] + q1[base];
  float y = proj_capped_simplex256(v, tid);
  q1[base] = v - y;
  p[base]  = y + q2[base];
}

__global__ __launch_bounds__(256) void dykstra_cols_k(float* __restrict__ p,
                                                      float* __restrict__ q2) {
  int j = blockIdx.x, b = blockIdx.y, tid = threadIdx.x;
  long base = (long)b * TT * TT + (long)tid * TT + j;
  float v = p[base];
  float y = proj_capped_simplex256(v, tid);
  q2[base] = v - y;
  p[base]  = y;
}

__global__ void z_finalize_k(const float* __restrict__ p,
                             float* __restrict__ zout,
                             unsigned short* __restrict__ zb, long n) {
  long i = (long)blockIdx.x * 256 + threadIdx.x;
  if (i >= n) return;
  float v = p[i];
  zout[i] = v;
  zb[i]   = f32_to_bf16(v);
}

__global__ __launch_bounds__(256) void pool_out_k(const float* __restrict__ c1,
                                                  const float* __restrict__ c2,
                                                  const float* __restrict__ Wo,
                                                  const float* __restrict__ bo,
                                                  float* __restrict__ out) {
  __shared__ float rep[REP];
  __shared__ float red[256];
  int b = blockIdx.x, tid = threadIdx.x;
  for (int f = tid; f < ENC; f += 256) {
    float s1 = 0.f, m1 = -3.4e38f, s2 = 0.f, m2 = -3.4e38f;
    for (int t = 0; t < TT; ++t) {
      float a = c1[((long)b * TT + t) * ENC + f];
      s1 += a; m1 = fmaxf(m1, a);
      float c = c2[((long)b * TT + t) * ENC + f];
      s2 += c; m2 = fmaxf(m2, c);
    }
    rep[f]           = s1 * (1.0f / TT);
    rep[ENC + f]     = m1;
    rep[2 * ENC + f] = s2 * (1.0f / TT);
    rep[3 * ENC + f] = m2;
  }
  __syncthreads();
  float acc = 0.f;
  for (int k = tid; k < REP; k += 256) acc += rep[k] * Wo[k];
  red[tid] = acc; __syncthreads();
  for (int s = 128; s > 0; s >>= 1) {
    if (tid < s) red[tid] += red[tid + s];
    __syncthreads();
  }
  if (tid == 0) out[b] = 1.0f / (1.0f + expf(-(red[0] + bo[0])));
}

// ---------------------------------------------------------------------------
// Host orchestration
// ---------------------------------------------------------------------------
struct GemmArgs {
  const unsigned short* A; const unsigned short* B;
  const float* bias; const float* add;
  float* C; unsigned short* Cb16;
  int M, N, Kp;
  long sa_m, sa_k, a_bs, sb_k, sb_n, b_bs, c_row, c_bs, add_row;
  int relu, accum, batch;
};

template<int AM, int BM>
static inline void launch_gemm(hipStream_t stream, const GemmArgs& a) {
  dim3 g((unsigned)((a.N + 31) / 32), (unsigned)((a.M + 63) / 64), (unsigned)a.batch);
  wmma_gemm_t<AM, BM><<<g, 256, 0, stream>>>(
      a.A, a.B, a.bias, a.add, a.C, a.Cb16, a.M, a.N, a.Kp,
      a.sa_m, a.sa_k, a.a_bs, a.sb_k, a.sb_n, a.b_bs,
      a.c_row, a.c_bs, a.add_row, a.relu, a.accum);
}

extern "C" void kernel_launch(void* const* d_in, const int* in_sizes, int n_in,
                              void* d_out, int out_size, void* d_ws, size_t ws_size,
                              hipStream_t stream) {
  (void)in_sizes; (void)n_in; (void)out_size; (void)ws_size;

  const int*   x1     = (const int*)  d_in[0];
  const int*   x2     = (const int*)  d_in[1];
  const float* embed  = (const float*)d_in[4];
  const float* Wih_cf = (const float*)d_in[5];
  const float* Whh_cf = (const float*)d_in[6];
  const float* b_cf   = (const float*)d_in[7];
  const float* Wih_cb = (const float*)d_in[8];
  const float* Whh_cb = (const float*)d_in[9];
  const float* b_cb   = (const float*)d_in[10];
  const float* Wp1    = (const float*)d_in[11];
  const float* bp1    = (const float*)d_in[12];
  const float* Wp2    = (const float*)d_in[13];
  const float* bp2    = (const float*)d_in[14];
  const float* Wih_mf = (const float*)d_in[15];
  const float* Whh_mf = (const float*)d_in[16];
  const float* b_mf   = (const float*)d_in[17];
  const float* Wih_mb = (const float*)d_in[18];
  const float* Whh_mb = (const float*)d_in[19];
  const float* b_mb   = (const float*)d_in[20];
  const float* Wo     = (const float*)d_in[21];
  const float* bo     = (const float*)d_in[22];
  float* out = (float*)d_out;

  size_t off = 0;
  auto alloc = [&](size_t bytes) -> char* {
    off = (off + 255) & ~(size_t)255;
    char* p = (char*)d_ws + off;
    off += bytes;
    return p;
  };
  const long TB = (long)TT * BB;  // 16384

  unsigned short* e1_tm   = (unsigned short*)alloc(TB * EP * 2);
  unsigned short* e2_tm   = (unsigned short*)alloc(TB * EP * 2);
  unsigned short* wihcf   = (unsigned short*)alloc((long)G4 * EP * 2);
  unsigned short* wihcb   = (unsigned short*)alloc((long)G4 * EP * 2);
  unsigned short* whhcf   = (unsigned short*)alloc((long)G4 * HP * 2);
  unsigned short* whhcb   = (unsigned short*)alloc((long)G4 * HP * 2);
  unsigned short* wihmf   = (unsigned short*)alloc((long)G4 * HP * 2);
  unsigned short* wihmb   = (unsigned short*)alloc((long)G4 * HP * 2);
  unsigned short* whhmf   = (unsigned short*)alloc((long)G4 * HP * 2);
  unsigned short* whhmb   = (unsigned short*)alloc((long)G4 * HP * 2);
  unsigned short* wp1b    = (unsigned short*)alloc((long)HP * ENCP * 2);
  unsigned short* wp2b_lo = (unsigned short*)alloc((long)HP * ENCP * 2);
  unsigned short* wp2b_hi = (unsigned short*)alloc((long)HP * ENCP * 2);
  float*          xproj   = (float*)alloc(TB * G4 * 4);
  float*          gbuf    = (float*)alloc((long)BB * G4 * 4);
  unsigned short* hbf     = (unsigned short*)alloc((long)BB * HP * 2);
  float*          cst     = (float*)alloc((long)BB * HH * 4);
  float*          x1h     = (float*)alloc(TB * ENCP * 4);   // [B,T,ENCP]
  float*          x2h     = (float*)alloc(TB * ENCP * 4);
  unsigned short* x1hb    = (unsigned short*)alloc(TB * ENCP * 2);
  unsigned short* x2hb    = (unsigned short*)alloc(TB * ENCP * 2);
  float*          p       = (float*)alloc((long)BB * TT * TT * 4);
  float*          q1      = (float*)alloc((long)BB * TT * TT * 4);
  float*          q2      = (float*)alloc((long)BB * TT * TT * 4);
  unsigned short* zb      = (unsigned short*)alloc((long)BB * TT * TT * 2);
  float*          x1al    = (float*)alloc(TB * ENCP * 4);   // time-major
  float*          x2al    = (float*)alloc(TB * ENCP * 4);
  unsigned short* x1alb   = (unsigned short*)alloc(TB * ENCP * 2);
  unsigned short* x2alb   = (unsigned short*)alloc(TB * ENCP * 2);
  float*          x1c     = (float*)alloc(TB * HP * 4);     // time-major
  float*          x2c     = (float*)alloc(TB * HP * 4);
  unsigned short* x1cb    = (unsigned short*)alloc(TB * HP * 2);
  unsigned short* x2cb    = (unsigned short*)alloc(TB * HP * 2);
  float*          comp1   = (float*)alloc(TB * ENC * 4);    // [B,T,2H]
  float*          comp2   = (float*)alloc(TB * ENC * 4);

  auto cast_bf16 = [&](unsigned short* d, const float* s, long n) {
    cast_f32_bf16_k<<<dim3((unsigned)((n + 255) / 256)), 256, 0, stream>>>(d, s, n);
  };
  auto cast_pad = [&](unsigned short* d, const float* s, int sld, int soff,
                      int Nr, int Kc, int Np, int Kp2) {
    long n = (long)Np * Kp2;
    cast_pad2d_k<<<dim3((unsigned)((n + 255) / 256)), 256, 0, stream>>>(
        d, s, sld, soff, Nr, Kc, Np, Kp2);
  };
  auto zero_words = [&](void* pz, long words) {
    fill_u32_k<<<dim3((unsigned)((words + 255) / 256)), 256, 0, stream>>>(
        (unsigned int*)pz, words);
  };

  // ---- weight casts into padded bf16 --------------------------------------
  cast_pad(wihcf,   Wih_cf, EE,  0,   G4, EE,  G4, EP);
  cast_pad(wihcb,   Wih_cb, EE,  0,   G4, EE,  G4, EP);
  cast_pad(whhcf,   Whh_cf, HH,  0,   G4, HH,  G4, HP);
  cast_pad(whhcb,   Whh_cb, HH,  0,   G4, HH,  G4, HP);
  cast_pad(wihmf,   Wih_mf, HH,  0,   G4, HH,  G4, HP);
  cast_pad(wihmb,   Wih_mb, HH,  0,   G4, HH,  G4, HP);
  cast_pad(whhmf,   Whh_mf, HH,  0,   G4, HH,  G4, HP);
  cast_pad(whhmb,   Whh_mb, HH,  0,   G4, HH,  G4, HP);
  cast_pad(wp1b,    Wp1,    ENC, 0,   HH, ENC, HP, ENCP);
  cast_pad(wp2b_lo, Wp2,    G4,  0,   HH, ENC, HP, ENCP);
  cast_pad(wp2b_hi, Wp2,    G4,  ENC, HH, ENC, HP, ENCP);

  // ---- zero-fills that establish padding invariants -----------------------
  zero_words(x1h,   TB * ENCP);
  zero_words(x2h,   TB * ENCP);
  zero_words(x1alb, TB * ENCP / 2);
  zero_words(x2alb, TB * ENCP / 2);
  zero_words(x1cb,  TB * HP / 2);
  zero_words(x2cb,  TB * HP / 2);
  zero_words(q1, (long)BB * TT * TT);
  zero_words(q2, (long)BB * TT * TT);

  // ---- embedding gather (time-major, padded bf16) -------------------------
  {
    long n = TB * EP;
    embed_gather_k<<<dim3((unsigned)((n + 255) / 256)), 256, 0, stream>>>(e1_tm, embed, x1, n);
    embed_gather_k<<<dim3((unsigned)((n + 255) / 256)), 256, 0, stream>>>(e2_tm, embed, x2, n);
  }

  // ---- 8 LSTM recurrences -------------------------------------------------
  struct Rec {
    const unsigned short* in; int KinP;
    const unsigned short* wih; const unsigned short* whh; const float* bias;
    float* hseq; long seq_bstride; long seq_tpitch; int dir;
  };
  auto run_recs = [&](const Rec* recs, int nrec) {
    for (int r = 0; r < nrec; ++r) {
      const Rec& R = recs[r];
      // input projection: [T*B, KinP] @ [4H, KinP]^T -> xproj [T,B,4H]
      GemmArgs xa = { R.in, R.wih, nullptr, nullptr, xproj, nullptr,
                      (int)TB, G4, R.KinP,
                      (long)R.KinP, 1, 0,   1, (long)R.KinP, 0,
                      G4, 0, 0, 0, 0, 1 };
      launch_gemm<0, 0>(stream, xa);
      zero_words(hbf, (long)BB * HP / 2);
      zero_words(cst, (long)BB * HH);
      for (int s = 0; s < TT; ++s) {
        int t = (R.dir == 0) ? s : (TT - 1 - s);
        GemmArgs ga = { hbf, R.whh, R.bias, xproj + (long)t * BB * G4,
                        gbuf, nullptr,
                        BB, G4, HP,
                        HP, 1, 0,   1, HP, 0,
                        G4, 0, G4, 0, 0, 1 };
        launch_gemm<0, 0>(stream, ga);
        lstm_gate_k<<<dim3((BB * HH + 255) / 256), 256, 0, stream>>>(
            gbuf, cst, hbf, R.hseq,
            (long)t * R.seq_tpitch + (long)R.dir * HH, R.seq_bstride);
      }
    }
  };

  Rec ctx[4] = {
    { e1_tm, EP, wihcf, whhcf, b_cf, x1h, (long)TT * ENCP, ENCP, 0 },
    { e1_tm, EP, wihcb, whhcb, b_cb, x1h, (long)TT * ENCP, ENCP, 1 },
    { e2_tm, EP, wihcf, whhcf, b_cf, x2h, (long)TT * ENCP, ENCP, 0 },
    { e2_tm, EP, wihcb, whhcb, b_cb, x2h, (long)TT * ENCP, ENCP, 1 },
  };
  run_recs(ctx, 4);

  cast_bf16(x1hb, x1h, TB * ENCP);
  cast_bf16(x2hb, x2h, TB * ENCP);

  // ---- scores[b,i,j] = sum_f x1h[b,i,f]*x2h[b,j,f] ------------------------
  {
    GemmArgs a = { x1hb, x2hb, nullptr, nullptr, p, nullptr,
                   TT, TT, ENCP,
                   ENCP, 1, (long)TT * ENCP,   1, ENCP, (long)TT * ENCP,
                   TT, (long)TT * TT, 0, 0, 0, BB };
    launch_gemm<0, 0>(stream, a);
  }

  // ---- Dykstra ------------------------------------------------------------
  for (int it = 0; it < 10; ++it) {
    dykstra_rows_k<<<dim3(TT, BB), 256, 0, stream>>>(p, q1, q2);
    dykstra_cols_k<<<dim3(TT, BB), 256, 0, stream>>>(p, q2);
  }

  {
    long n = (long)BB * TT * TT;
    z_finalize_k<<<dim3((unsigned)((n + 255) / 256)), 256, 0, stream>>>(p, out, zb, n);
  }

  // ---- alignments (time-major outputs) ------------------------------------
  // x1_align[b,i,f] = sum_j z[b,i,j]*x2h[b,j,f]   (A k-contig, B n-contig)
  {
    GemmArgs a = { zb, x2hb, nullptr, nullptr, x1al, x1alb,
                   TT, ENC, TT,
                   TT, 1, (long)TT * TT,   ENCP, 1, (long)TT * ENCP,
                   (long)BB * ENCP, ENCP, 0, 0, 0, BB };
    launch_gemm<0, 1>(stream, a);
  }
  // x2_align[b,j,f] = sum_i z[b,i,j]*x1h[b,i,f]   (A m-contig, B n-contig)
  {
    GemmArgs a = { zb, x1hb, nullptr, nullptr, x2al, x2alb,
                   TT, ENC, TT,
                   1, TT, (long)TT * TT,   ENCP, 1, (long)TT * ENCP,
                   (long)BB * ENCP, ENCP, 0, 0, 0, BB };
    launch_gemm<1, 1>(stream, a);
  }

  // ---- x1_c = relu(x1_align @ Wp1^T + bp1) --------------------------------
  {
    GemmArgs a = { x1alb, wp1b, bp1, nullptr, x1c, x1cb,
                   (int)TB, HH, ENCP,
                   ENCP, 1, 0,   1, ENCP, 0,
                   HP, 0, 0, 1, 0, 1 };
    launch_gemm<0, 0>(stream, a);
  }

  // ---- x2_c = relu([x2_h | x2_align] @ Wp2^T + bp2): two passes -----------
  {
    GemmArgs a = { x2hb, wp2b_lo, nullptr, nullptr, x2c, nullptr,
                   TT, HH, ENCP,
                   ENCP, 1, (long)TT * ENCP,   1, ENCP, 0,
                   (long)BB * HP, HP, 0, 0, 0, BB };
    launch_gemm<0, 0>(stream, a);
  }
  {
    GemmArgs a = { x2alb, wp2b_hi, bp2, nullptr, x2c, x2cb,
                   TT, HH, ENCP,
                   (long)BB * ENCP, 1, ENCP,   1, ENCP, 0,
                   (long)BB * HP, HP, 0, 1, 1, BB };
    launch_gemm<0, 0>(stream, a);
  }

  // ---- matching BiLSTMs (comp pitch = ENC, unpadded) ----------------------
  Rec mat[4] = {
    { x1cb, HP, wihmf, whhmf, b_mf, comp1, (long)TT * ENC, ENC, 0 },
    { x1cb, HP, wihmb, whhmb, b_mb, comp1, (long)TT * ENC, ENC, 1 },
    { x2cb, HP, wihmf, whhmf, b_mf, comp2, (long)TT * ENC, ENC, 0 },
    { x2cb, HP, wihmb, whhmb, b_mb, comp2, (long)TT * ENC, ENC, 1 },
  };
  run_recs(mat, 4);

  // ---- pooling + output ---------------------------------------------------
  pool_out_k<<<dim3(BB), 256, 0, stream>>>(comp1, comp2, Wo, bo,
                                           out + (long)BB * TT * TT);
}
